// R_REN_36386962931991
// MI455X (gfx1250) — matrix operand order repaired and checked by reading, not compiled, not run on hardware
//
#include <hip/hip_runtime.h>
#include <math.h>

#define N_STATES 128
#define N_NL     64
#define N_IN     64
#define N_OUT    32
#define N_SYS    16
#define FILT     24
#define GAMMA_C  0.001f
#define Q_VAL    10.0f
#define BETA_C   100000.0f
#define EPS_C    0.001f

// ---------------- workspace layout (float offsets) ----------------
#define OFF_D22   0          // 32*64   = 2048
#define OFF_RCI   2048       // 64*64   = 4096
#define OFF_VR    6144       // 320*64  = 20480
#define OFF_VQ    26624      // 320*32  = 10240
#define OFF_T     36864      // 320*64  = 20480
#define OFF_H     57344      // 320*320 = 102400
#define OFF_E     159744     // 128*128 = 16384
#define OFF_EINV  176128     // 16384
#define OFF_WE    192512     // 128*256 = 32768
#define OFF_G     225280     // 32768
#define OFF_D11   258048     // 4096
#define OFF_RLAM  262144     // 64
#define OFF_WA    262208     // 64*256 bf16 -> 8192 floats
#define OFF_P     270400     // 160*256 bf16 -> 20480 floats

typedef __bf16 v16bf __attribute__((ext_vector_type(16)));
typedef float  v8f   __attribute__((ext_vector_type(8)));
typedef unsigned int u32x4 __attribute__((ext_vector_type(4)));

union FragU { u32x4 u4[2]; v16bf v; };

__device__ __forceinline__ unsigned short f2bf(float x) {
    unsigned int u = __float_as_uint(x);
    unsigned int r = u + 0x7FFFu + ((u >> 16) & 1u);   // round-to-nearest-even
    return (unsigned short)(r >> 16);
}
__device__ __forceinline__ unsigned int packbf(float a, float b) {
    return (unsigned int)f2bf(a) | ((unsigned int)f2bf(b) << 16);
}

__device__ __forceinline__ float Rdiag(int j) {
    if (j < 16) return BETA_C;
    if (j < 39) return GAMMA_C;
    if (j == 39) return BETA_C;
    return GAMMA_C;
}

// ---- CDNA5 async global->LDS copy (ASYNCcnt-tracked) ----
// GVS mode: mem = SGPR64(base) + VGPR32(goff);  LDS dest = VGPR32(ldsOff)
__device__ __forceinline__ void async_b128(unsigned ldsOff, unsigned gOff,
                                           const void* gbase) {
    asm volatile("global_load_async_to_lds_b128 %0, %1, %2"
                 :: "v"(ldsOff), "v"(gOff),
                    "s"((unsigned long long)(size_t)gbase)
                 : "memory");
}
__device__ __forceinline__ void wait_async0() {
#if __has_builtin(__builtin_amdgcn_s_wait_asynccnt)
    __builtin_amdgcn_s_wait_asynccnt(0);
#else
    asm volatile("s_wait_asynccnt 0" ::: "memory");
#endif
}

// ============ derived small-matrix setup (one block, 256 thr) ============
__global__ void k_setup(const float* __restrict__ X3, const float* __restrict__ Y3,
                        const float* __restrict__ Z3, const float* __restrict__ C2,
                        const float* __restrict__ D21, const float* __restrict__ D12,
                        const float* __restrict__ B2, float* __restrict__ ws)
{
    __shared__ float MS[32 * 32];
    __shared__ float Aug32[32 * 64];
    __shared__ float D22S[32 * 64];
    __shared__ float RAug[64 * 128];
    __shared__ float colP[64];
    __shared__ float pivS;
    const int tid = threadIdx.x;

    for (int e = tid; e < 1024; e += 256) {
        int i = e >> 5, j = e & 31;
        float s = (i == j) ? EPS_C : 0.0f;
        for (int k = 0; k < 32; ++k)
            s += X3[k * 32 + i] * X3[k * 32 + j] + Z3[k * 32 + i] * Z3[k * 32 + j];
        s += Y3[i * 32 + j] - Y3[j * 32 + i];
        MS[e] = s;
    }
    __syncthreads();
    for (int e = tid; e < 2048; e += 256) {
        int i = e >> 6, j = e & 63;
        Aug32[e] = (j < 32) ? (MS[i * 32 + j] + ((i == j) ? 1.0f : 0.0f))
                            : ((j - 32 == i) ? 1.0f : 0.0f);
    }
    __syncthreads();
    for (int p = 0; p < 32; ++p) {
        if (tid == 0) pivS = 1.0f / Aug32[p * 64 + p];
        __syncthreads();
        for (int c = tid; c < 64; c += 256) Aug32[p * 64 + c] *= pivS;
        __syncthreads();
        for (int r = tid; r < 32; r += 256) colP[r] = (r == p) ? 0.0f : Aug32[r * 64 + p];
        __syncthreads();
        for (int e = tid; e < 2048; e += 256) {
            int r = e >> 6, c = e & 63;
            Aug32[e] -= colP[r] * Aug32[p * 64 + c];
        }
        __syncthreads();
    }
    for (int e = tid; e < 2048; e += 256) {
        int i = e >> 6, j = e & 63;
        float s = 0.0f;
        if (j < 32) {
            for (int k = 0; k < 32; ++k)
                s += Aug32[i * 64 + 32 + k] * (((k == j) ? 1.0f : 0.0f) - MS[k * 32 + j]);
        } else {
            for (int k = 0; k < 32; ++k)
                s += Aug32[i * 64 + 32 + k] * Z3[(j - 32) * 32 + k];
            s *= -2.0f;
        }
        float d = s * sqrtf(Rdiag(j) * 0.1f);
        D22S[e] = d;
        ws[OFF_D22 + e] = d;
    }
    __syncthreads();
    for (int e = tid; e < 8192; e += 256) {
        int i = e >> 7, j = e & 127;
        float v;
        if (j < 64) {
            float s = 0.0f;
            for (int k = 0; k < 32; ++k) s += D22S[k * 64 + i] * D22S[k * 64 + j];
            v = ((i == j) ? Rdiag(i) : 0.0f) - 10.0f * s;
        } else v = (j - 64 == i) ? 1.0f : 0.0f;
        RAug[e] = v;
    }
    __syncthreads();
    for (int p = 0; p < 64; ++p) {
        if (tid == 0) pivS = 1.0f / RAug[p * 128 + p];
        __syncthreads();
        for (int c = tid; c < 128; c += 256) RAug[p * 128 + c] *= pivS;
        __syncthreads();
        for (int r = tid; r < 64; r += 256) colP[r] = (r == p) ? 0.0f : RAug[r * 128 + p];
        __syncthreads();
        for (int e = tid; e < 8192; e += 256) {
            int r = e >> 7, c = e & 127;
            RAug[e] -= colP[r] * RAug[p * 128 + c];
        }
        __syncthreads();
    }
    for (int e = tid; e < 4096; e += 256) {
        int i = e >> 6, j = e & 63;
        ws[OFF_RCI + e] = RAug[i * 128 + 64 + j];
    }
    for (int e = tid; e < 320 * 64; e += 256) {
        int i = e >> 6, j = e & 63;
        float v;
        if (i < 128) {
            float s = 0.0f;
            for (int k = 0; k < 32; ++k) s += C2[k * 128 + i] * D22S[k * 64 + j];
            v = -10.0f * s;
        } else if (i < 192) {
            int ii = i - 128;
            float s = 0.0f;
            for (int k = 0; k < 32; ++k) s += D21[k * 64 + ii] * D22S[k * 64 + j];
            v = -10.0f * s - D12[ii * 64 + j];
        } else v = B2[(i - 192) * 64 + j];
        ws[OFF_VR + e] = v;
    }
    for (int e = tid; e < 320 * 32; e += 256) {
        int i = e >> 5, j = e & 31;
        float v;
        if (i < 128) v = C2[j * 128 + i];
        else if (i < 192) v = D21[j * 64 + (i - 128)];
        else v = 0.0f;
        ws[OFF_VQ + e] = v;
    }
}

// ============ generic f32 GEMM: C = alpha*opA(A)@opB(B) (+C) ============
__global__ void k_mm(const float* __restrict__ A, const float* __restrict__ B,
                     float* __restrict__ C, int M, int N, int K,
                     int ta, int tb, float alpha, int accum)
{
    int idx = blockIdx.x * blockDim.x + threadIdx.x;
    if (idx >= M * N) return;
    int i = idx / N, j = idx % N;
    float s = 0.0f;
    for (int k = 0; k < K; ++k) {
        float a = ta ? A[k * M + i] : A[i * K + k];
        float b = tb ? B[j * K + k] : B[k * N + j];
        s += a * b;
    }
    C[idx] = accum ? (C[idx] + alpha * s) : (alpha * s);
}

// ============ extract E, D11, rLam, We (f32) and Wa (bf16) from H ============
__global__ void k_extract(const float* __restrict__ Y1, const float* __restrict__ B2,
                          const float* __restrict__ D12, float* __restrict__ ws)
{
    const float* H = ws + OFF_H;
    const int tid = threadIdx.x;
    for (int e = tid; e < 128 * 128; e += 256) {
        int i = e >> 7, j = e & 127;
        float v = 0.5f * (H[i * 320 + j] + H[(192 + i) * 320 + 192 + j]
                          + Y1[i * 128 + j] - Y1[j * 128 + i]);
        if (i == j) v += EPS_C;
        ws[OFF_E + e] = v;
    }
    for (int e = tid; e < 64 * 64; e += 256) {
        int i = e >> 6, j = e & 63;
        ws[OFF_D11 + e] = (j < i) ? -H[(128 + i) * 320 + 128 + j] : 0.0f;
    }
    if (tid < 64) ws[OFF_RLAM + tid] = 2.0f / (H[(128 + tid) * 320 + 128 + tid] + EPS_C);
    for (int e = tid; e < 128 * 256; e += 256) {
        int i = e >> 8, c = e & 255;
        float v;
        if (c < 128) v = H[(192 + i) * 320 + c];
        else if (c < 192) v = B2[i * 64 + (c - 128)];
        else v = H[(192 + i) * 320 + 128 + (c - 192)];
        ws[OFF_WE + e] = v;
    }
    unsigned short* WaU = (unsigned short*)(ws + OFF_WA);
    for (int e = tid; e < 64 * 256; e += 256) {
        int i = e >> 8, c = e & 255;
        float v;
        if (c < 128) v = -H[(128 + i) * 320 + c];
        else if (c < 192) v = D12[i * 64 + (c - 128)];
        else v = 0.0f;
        WaU[e] = f2bf(v);
    }
}

// ============ 128x128 Gauss-Jordan inverse in LDS ============
__global__ void k_inv128(const float* __restrict__ E, float* __restrict__ Einv)
{
    extern __shared__ float Aug[];   // 128*256 floats
    __shared__ float colP[128];
    __shared__ float pivS;
    const int tid = threadIdx.x;     // 256 threads
    for (int e = tid; e < 128 * 256; e += 256) {
        int i = e >> 8, j = e & 255;
        Aug[e] = (j < 128) ? E[i * 128 + j] : ((j - 128 == i) ? 1.0f : 0.0f);
    }
    __syncthreads();
    for (int p = 0; p < 128; ++p) {
        if (tid == 0) pivS = 1.0f / Aug[p * 256 + p];
        __syncthreads();
        Aug[p * 256 + tid] *= pivS;
        __syncthreads();
        if (tid < 128) colP[tid] = (tid == p) ? 0.0f : Aug[tid * 256 + p];
        __syncthreads();
        for (int e = tid; e < 128 * 256; e += 256) {
            int r = e >> 8, c = e & 255;
            Aug[e] -= colP[r] * Aug[p * 256 + c];
        }
        __syncthreads();
    }
    for (int e = tid; e < 128 * 128; e += 256) {
        int i = e >> 7, j = e & 127;
        Einv[e] = Aug[i * 256 + 128 + j];
    }
}

// ============ pack P = [[C2|D22|D21]; G] as bf16 (160 x 256) ============
__global__ void k_pack(const float* __restrict__ C2, const float* __restrict__ D21,
                       float* __restrict__ ws)
{
    int e = blockIdx.x * blockDim.x + threadIdx.x;
    if (e >= 160 * 256) return;
    int row = e >> 8, c = e & 255;
    float v;
    if (row < 32) {
        if (c < 128) v = C2[row * 128 + c];
        else if (c < 192) v = ws[OFF_D22 + row * 64 + (c - 128)];
        else v = D21[row * 64 + (c - 192)];
    } else v = ws[OFF_G + (row - 32) * 256 + c];
    ((unsigned short*)(ws + OFF_P))[e] = f2bf(v);
}

// ============ batched REN forward: async-LDS staging + bf16 WMMA ============
#define RS   264     // LDS row stride in bf16 (132 dwords; 132%64=4 -> conflict-free)
#define RSD  132
#define RSB  528     // row stride in bytes

// LDS byte offsets inside dynamic shared block
#define L_VT    0u
#define L_WA    (128u * RSB)              // 67584
#define L_PS    (L_WA + 64u * RSB)        // 101376
#define L_D11   (L_PS + 160u * RSB)       // 185856
#define L_RLAM  (L_D11 + 16384u)          // 202240
#define L_SCR   (L_RLAM + 256u)           // 202496 (96KB union: f32 stage / a-buffer)
#define L_TOTAL (L_SCR + 98304u)          // 300800 bytes

__global__ __launch_bounds__(256) void k_batch(
    const float* __restrict__ state, const float* __restrict__ inpt,
    const float* __restrict__ ws, float* __restrict__ out, int Btot)
{
    extern __shared__ char smem[];
    unsigned short* Vt   = (unsigned short*)(smem + L_VT);
    unsigned short* WaS  = (unsigned short*)(smem + L_WA);
    unsigned short* Ps   = (unsigned short*)(smem + L_PS);
    float*          D11s = (float*)(smem + L_D11);
    float*          rLamS= (float*)(smem + L_RLAM);
    float*          aw   = (float*)(smem + L_SCR);       // 128 x 65 f32 (after staging)
    const float*    Sst  = (const float*)(smem + L_SCR);           // 128x128 f32 stage
    const float*    Sin  = (const float*)(smem + L_SCR + 65536u);  // 128x64  f32 stage

    const int tid = threadIdx.x;            // 256 threads = 8 waves
    const int bbase = blockIdx.x * 128;
    const int wv = tid >> 5, lane = tid & 31, lm = lane & 15, lh = lane >> 4;
    const unsigned ldsBase = (unsigned)(size_t)smem;   // generic->LDS is low-32 truncate
    const unsigned lb = (unsigned)lane * 16u;
    const bool full = (bbase + 128 <= Btot);

    // ---- async-stage weights: one wave-wide B128 per 512B row ----
    const void* WaG = (const void*)(ws + OFF_WA);
    const void* PG  = (const void*)(ws + OFF_P);
    const void* DG  = (const void*)(ws + OFF_D11);
    for (int r = wv; r < 64; r += 8)
        async_b128(ldsBase + L_WA + (unsigned)r * RSB + lb, (unsigned)r * 512u + lb, WaG);
    for (int r = wv; r < 160; r += 8)
        async_b128(ldsBase + L_PS + (unsigned)r * RSB + lb, (unsigned)r * 512u + lb, PG);
    for (int u = wv; u < 32; u += 8)
        async_b128(ldsBase + L_D11 + (unsigned)u * 512u + lb, (unsigned)u * 512u + lb, DG);
    // ---- async-stage activations (f32) when the block is full ----
    if (full) {
        const void* SB = (const void*)(state + (size_t)bbase * 128);
        const void* IB = (const void*)(inpt  + (size_t)bbase * 64);
        for (int u = wv; u < 128; u += 8)
            async_b128(ldsBase + L_SCR + (unsigned)u * 512u + lb, (unsigned)u * 512u + lb, SB);
        for (int u = wv; u < 64; u += 8)
            async_b128(ldsBase + L_SCR + 65536u + (unsigned)u * 512u + lb,
                       (unsigned)u * 512u + lb, IB);
    }
    if (tid < 64) rLamS[tid] = ws[OFF_RLAM + tid];
    wait_async0();
    __syncthreads();

    // ---- convert staged f32 -> bf16 V tile = [state | inpt | w(=0)] ----
    unsigned int* Vd = (unsigned int*)Vt;
    if (full) {
        for (int e = tid; e < 128 * 64; e += 256) {
            int row = e >> 6, p = e & 63;
            float2 s = ((const float2*)Sst)[row * 64 + p];
            Vd[row * RSD + p] = packbf(s.x, s.y);
        }
        for (int e = tid; e < 128 * 32; e += 256) {
            int row = e >> 5, p = e & 31;
            float2 s = ((const float2*)Sin)[row * 32 + p];
            Vd[row * RSD + 64 + p] = packbf(s.x, s.y);
            Vd[row * RSD + 96 + p] = 0u;
        }
    } else {
        for (int e = tid; e < 128 * 64; e += 256) {
            int row = e >> 6, p = e & 63;
            int gb = bbase + row;
            float2 s; s.x = 0.0f; s.y = 0.0f;
            if (gb < Btot) s = ((const float2*)(state + (size_t)gb * 128))[p];
            Vd[row * RSD + p] = packbf(s.x, s.y);
        }
        for (int e = tid; e < 128 * 32; e += 256) {
            int row = e >> 5, p = e & 31;
            int gb = bbase + row;
            float2 s; s.x = 0.0f; s.y = 0.0f;
            if (gb < Btot) s = ((const float2*)(inpt + (size_t)gb * 64))[p];
            Vd[row * RSD + 64 + p] = packbf(s.x, s.y);
            Vd[row * RSD + 96 + p] = 0u;
        }
    }
    __syncthreads();

    // ---- phase 1: a = V @ Wa^T (128x64); wave wv owns row-block mt = wv ----
    {
        const unsigned int* ar = (const unsigned int*)(Vt + (wv * 16 + lm) * RS);
        FragU fa[8];
        #pragma unroll
        for (int kb = 0; kb < 8; ++kb) {
            fa[kb].u4[0] = *(const u32x4*)(ar + kb * 16 + lh * 4);
            fa[kb].u4[1] = *(const u32x4*)(ar + kb * 16 + 8 + lh * 4);
        }
        for (int nt = 0; nt < 4; ++nt) {
            v8f c;
            #pragma unroll
            for (int r = 0; r < 8; ++r) c[r] = 0.0f;
            const unsigned int* br = (const unsigned int*)(WaS + (nt * 16 + lm) * RS);
            #pragma unroll
            for (int kb = 0; kb < 8; ++kb) {
                FragU fb;
                fb.u4[0] = *(const u32x4*)(br + kb * 16 + lh * 4);
                fb.u4[1] = *(const u32x4*)(br + kb * 16 + 8 + lh * 4);
                c = __builtin_amdgcn_wmma_f32_16x16x32_bf16(false, fa[kb].v, false, fb.v,
                                                            (short)0, c, false, false);
            }
            #pragma unroll
            for (int r = 0; r < 8; ++r)
                aw[(wv * 16 + lh * 8 + r) * 65 + nt * 16 + lm] = c[r];
        }
    }
    __syncthreads();

    // ---- phase 2: tanh triangular recurrence, w in VGPRs (full unroll) ----
    if (tid < 128) {
        const float* __restrict__ arow = aw + tid * 65;
        float wreg[64];
        #pragma unroll
        for (int i = 0; i < 64; ++i) {
            float v0 = arow[i], v1 = 0.0f, v2 = 0.0f, v3 = 0.0f;
            const float4* drow = (const float4*)(D11s + i * 64);
            #pragma unroll
            for (int jq = 0; jq < i / 4; ++jq) {
                float4 d = drow[jq];
                v0 += wreg[4 * jq + 0] * d.x;
                v1 += wreg[4 * jq + 1] * d.y;
                v2 += wreg[4 * jq + 2] * d.z;
                v3 += wreg[4 * jq + 3] * d.w;
            }
            #pragma unroll
            for (int j = (i / 4) * 4; j < i; ++j) v0 += wreg[j] * D11s[i * 64 + j];
            float wi = tanhf(((v0 + v1) + (v2 + v3)) * rLamS[i]);
            wreg[i] = wi;
            Vt[tid * RS + 192 + i] = f2bf(wi);
        }
    }
    __syncthreads();

    // ---- phase 3: [out | state+] = V @ P^T (128x160); mt = wv, nt 0..9 ----
    {
        float* outSP = out + (size_t)Btot * 32;
        const unsigned int* ar = (const unsigned int*)(Vt + (wv * 16 + lm) * RS);
        FragU fa[8];
        #pragma unroll
        for (int kb = 0; kb < 8; ++kb) {      // reload: w block now filled
            fa[kb].u4[0] = *(const u32x4*)(ar + kb * 16 + lh * 4);
            fa[kb].u4[1] = *(const u32x4*)(ar + kb * 16 + 8 + lh * 4);
        }
        for (int nt = 0; nt < 10; ++nt) {
            v8f c;
            #pragma unroll
            for (int r = 0; r < 8; ++r) c[r] = 0.0f;
            const unsigned int* br = (const unsigned int*)(Ps + (nt * 16 + lm) * RS);
            #pragma unroll
            for (int kb = 0; kb < 8; ++kb) {
                FragU fb;
                fb.u4[0] = *(const u32x4*)(br + kb * 16 + lh * 4);
                fb.u4[1] = *(const u32x4*)(br + kb * 16 + 8 + lh * 4);
                c = __builtin_amdgcn_wmma_f32_16x16x32_bf16(false, fa[kb].v, false, fb.v,
                                                            (short)0, c, false, false);
            }
            #pragma unroll
            for (int r = 0; r < 8; ++r) {
                int m = wv * 16 + lh * 8 + r;
                int gb = bbase + m;
                if (gb >= Btot) continue;
                int n = nt * 16 + lm;
                if (n < 32) out[(size_t)gb * 32 + n] = c[r];
                else        outSP[(size_t)gb * 128 + (n - 32)] = c[r];
            }
        }
    }
}

extern "C" void kernel_launch(void* const* d_in, const int* in_sizes, int n_in,
                              void* d_out, int out_size, void* d_ws, size_t ws_size,
                              hipStream_t stream) {
    const float* inpt  = (const float*)d_in[0];
    const float* state = (const float*)d_in[1];
    const float* X     = (const float*)d_in[2];
    const float* Y1    = (const float*)d_in[3];
    const float* X3    = (const float*)d_in[4];
    const float* Y3    = (const float*)d_in[5];
    const float* Z3    = (const float*)d_in[6];
    const float* B2    = (const float*)d_in[7];
    const float* C2    = (const float*)d_in[8];
    const float* D21   = (const float*)d_in[9];
    const float* D12   = (const float*)d_in[10];
    float* ws  = (float*)d_ws;
    float* out = (float*)d_out;
    const int B = in_sizes[0] / N_IN;

    k_setup<<<1, 256, 0, stream>>>(X3, Y3, Z3, C2, D21, D12, B2, ws);
    k_mm<<<(320 * 64 + 255) / 256, 256, 0, stream>>>(ws + OFF_VR, ws + OFF_RCI, ws + OFF_T,
                                                     320, 64, 64, 0, 0, 1.0f, 0);
    k_mm<<<(320 * 320 + 255) / 256, 256, 0, stream>>>(X, X, ws + OFF_H,
                                                      320, 320, 320, 1, 0, 1.0f, 0);
    k_mm<<<(320 * 320 + 255) / 256, 256, 0, stream>>>(ws + OFF_T, ws + OFF_VR, ws + OFF_H,
                                                      320, 320, 64, 0, 1, 1.0f, 1);
    k_mm<<<(320 * 320 + 255) / 256, 256, 0, stream>>>(ws + OFF_VQ, ws + OFF_VQ, ws + OFF_H,
                                                      320, 320, 32, 0, 1, 10.0f, 1);
    k_extract<<<1, 256, 0, stream>>>(Y1, B2, D12, ws);
    k_inv128<<<1, 256, 128 * 256 * 4, stream>>>(ws + OFF_E, ws + OFF_EINV);
    k_mm<<<(128 * 256 + 255) / 256, 256, 0, stream>>>(ws + OFF_EINV, ws + OFF_WE, ws + OFF_G,
                                                      128, 256, 128, 0, 0, 1.0f, 0);
    k_pack<<<(160 * 256 + 255) / 256, 256, 0, stream>>>(C2, D21, ws);

    const int gridB = (B + 127) / 128;
    k_batch<<<gridB, 256, L_TOTAL, stream>>>(state, inpt, ws, out, B);
}
